// Decoder_37091337568880
// MI455X (gfx1250) — matrix operand order repaired and checked by reading, not compiled, not run on hardware
//
#include <hip/hip_runtime.h>
#include <hip/hip_bf16.h>

typedef __attribute__((ext_vector_type(16))) __bf16 v16bf;
typedef __attribute__((ext_vector_type(8)))  float  v8f;

#define B_    32
#define T_    128
#define U_    64
#define G_    256        // 4U
#define E_    512
#define V_    32000
#define TENC_ 128
#define M_ALL (B_*T_)    // 4096
#define NT_   (V_/16)    // 2000 N-tiles in the projection

// ---- WMMA bf16 fragment loaders (contiguous k = 16*half + e, same order in A and B) ----

// A matrix 16x32, row-major source: rowk0 points at A[m][k0], m = lane&15.
__device__ __forceinline__ v16bf ld_a_row(const float* rowk0, int half) {
  v16bf a;
#pragma unroll
  for (int e = 0; e < 16; ++e) {
    int k = half * 16 + e;
    a[e] = (__bf16)rowk0[k];
  }
  return a;
}

// B matrix 32x16 from row-major B[k][n] with row stride ldn; bk0n0 = &B[k0][n0].
__device__ __forceinline__ v16bf ld_b_strided(const float* bk0n0, int ldn, int lane) {
  int n = lane & 15, half = lane >> 4;
  v16bf b;
#pragma unroll
  for (int e = 0; e < 16; ++e)
    b[e] = (__bf16)bk0n0[(half * 16 + e) * ldn + n];
  return b;
}

// B matrix 32x16 where source is K-contiguous per column (B^T row-major).
__device__ __forceinline__ v16bf ld_b_rowmajor(const float* rowk0, int half) {
  v16bf b;
#pragma unroll
  for (int e = 0; e < 16; ++e)
    b[e] = (__bf16)rowk0[half * 16 + e];
  return b;
}

__device__ __forceinline__ v8f wmma_bf16(v16bf a, v16bf b, v8f c) {
  return __builtin_amdgcn_wmma_f32_16x16x32_bf16(false, a, false, b, (short)0, c,
                                                 false, false);
}

__device__ __forceinline__ float sigm(float x) { return 1.0f / (1.0f + __expf(-x)); }

// ---------------- Kernel 0: pack Wd into WMMA-B fragment order (bf16) -----------
// Layout: frag index f = (nt*4 + kk)*32 + lane ; 16 bf16 contiguous per lane.
// Value: pw[f][e] = Wd[(kk*32 + half*16 + e)*V + nt*16 + (lane&15)]
__global__ void __launch_bounds__(256) k_pack_wd(const float* __restrict__ Wd,
                                                 __bf16* __restrict__ pw) {
  int gt   = blockIdx.x * 256 + threadIdx.x;   // 0 .. NT_*4*32 - 1
  int lane = gt & 31;
  int frag = gt >> 5;                          // 0..7999
  int kk   = frag & 3;
  int nt   = frag >> 2;
  int n    = nt * 16 + (lane & 15);
  int half = lane >> 4;
  v16bf b;
#pragma unroll
  for (int e = 0; e < 16; ++e)
    b[e] = (__bf16)Wd[(long)(kk * 32 + half * 16 + e) * V_ + n];
  *(v16bf*)(pw + (long)gt * 16) = b;           // 32B contiguous per lane
}

// ---------------- Kernel 1: xz = emb[dec] @ Wx + bias  (M=4096,N=256,K=512) ----
__global__ void __launch_bounds__(256) k_embed_gemm(
    const int* __restrict__ dec, const float* __restrict__ emb,
    const float* __restrict__ Wx, const float* __restrict__ bias,
    float* __restrict__ xz) {
  int tid  = threadIdx.x;
  int lane = tid & 31, wave = tid >> 5;   // 8 waves
  int half = lane >> 4;
  int tileM = blockIdx.y * 16;
  int tileN = (blockIdx.x * 8 + wave) * 16;

  int mA = tileM + (lane & 15);
  const float* arow = emb + (long)dec[mA] * E_;

  v8f acc = {};
  for (int k0 = 0; k0 < E_; k0 += 32) {
    v16bf a = ld_a_row(arow + k0, half);
    v16bf b = ld_b_strided(Wx + k0 * G_ + tileN, G_, lane);
    acc = wmma_bf16(a, b, acc);
  }
  int n = tileN + (lane & 15);
  float bn = bias[n];
#pragma unroll
  for (int r = 0; r < 8; ++r) {
    int m = tileM + r + 8 * half;
    xz[(long)m * G_ + n] = acc[r] + bn;
  }
}

// ---------------- Kernel 2: LSTM scan, one 32-wave workgroup --------------------
__global__ void __launch_bounds__(1024) k_lstm(
    const float* __restrict__ xz, const float* __restrict__ Wh,
    const float* __restrict__ h0, const float* __restrict__ c0,
    float* __restrict__ seq, float* __restrict__ hT, float* __restrict__ cT) {
  __shared__ float h_lds[B_][U_];
  __shared__ float c_lds[B_][U_];
  __shared__ float z_lds[B_][G_];

  int tid  = threadIdx.x;
  int lane = tid & 31, wave = tid >> 5;   // 0..31
  int half = lane >> 4;
  int tileM = (wave >> 4) * 16;           // batch dim tile
  int tileN = (wave & 15) * 16;           // gate dim tile

  for (int idx = tid; idx < B_ * U_; idx += 1024) {
    h_lds[idx / U_][idx % U_] = h0[idx];
    c_lds[idx / U_][idx % U_] = c0[idx];
  }

  v16bf wb0 = ld_b_strided(Wh + 0  * G_ + tileN, G_, lane);
  v16bf wb1 = ld_b_strided(Wh + 32 * G_ + tileN, G_, lane);

  __syncthreads();

  int nC = tileN + (lane & 15);
  int mA = tileM + (lane & 15);

  for (int t = 0; t < T_; ++t) {
    v8f acc;
#pragma unroll
    for (int r = 0; r < 8; ++r) {
      int b = tileM + r + 8 * half;
      acc[r] = xz[((long)b * T_ + t) * G_ + nC];
    }
    v16bf a0 = ld_a_row(&h_lds[mA][0], half);
    v16bf a1 = ld_a_row(&h_lds[mA][32], half);
    acc = wmma_bf16(a0, wb0, acc);
    acc = wmma_bf16(a1, wb1, acc);
#pragma unroll
    for (int r = 0; r < 8; ++r)
      z_lds[tileM + r + 8 * half][nC] = acc[r];
    __syncthreads();

#pragma unroll
    for (int q = 0; q < 2; ++q) {
      int idx = tid + q * 1024;
      int b = idx >> 6, u = idx & 63;
      float i = sigm(z_lds[b][u]);
      float f = sigm(z_lds[b][U_ + u]);
      float g = tanhf(z_lds[b][2 * U_ + u]);
      float o = sigm(z_lds[b][3 * U_ + u]);
      float cn = f * c_lds[b][u] + i * g;
      float hn = o * tanhf(cn);
      c_lds[b][u] = cn;
      h_lds[b][u] = hn;
      seq[((long)b * T_ + t) * U_ + u] = hn;
    }
    __syncthreads();
  }

  for (int idx = tid; idx < B_ * U_; idx += 1024) {
    hT[idx] = h_lds[idx / U_][idx % U_];
    cT[idx] = c_lds[idx / U_][idx % U_];
  }
}

// ---------------- Kernel 3: attention per batch ---------------------------------
__global__ void __launch_bounds__(256) k_attn(
    const float* __restrict__ enc, const float* __restrict__ h0,
    const float* __restrict__ seq, float* __restrict__ ctx) {
  __shared__ float sc[T_][TENC_];          // 64 KB
  int b    = blockIdx.x;
  int tid  = threadIdx.x;
  int lane = tid & 31, wave = tid >> 5;
  int half = lane >> 4;
  const float* encB = enc + (long)b * TENC_ * U_;

  // scores = kv @ enc^T  (M=T, N=T_ENC, K=U)
  {
    int mt = wave * 16 + (lane & 15);
    const float* arow = (mt == 0) ? (h0 + b * U_)
                                  : (seq + ((long)b * T_ + (mt - 1)) * U_);
    v16bf a0 = ld_a_row(arow, half);
    v16bf a1 = ld_a_row(arow + 32, half);
    for (int s0 = 0; s0 < TENC_; s0 += 16) {
      const float* brow = encB + (s0 + (lane & 15)) * U_;
      v8f acc = {};
      acc = wmma_bf16(a0, ld_b_rowmajor(brow, half), acc);
      acc = wmma_bf16(a1, ld_b_rowmajor(brow + 32, half), acc);
#pragma unroll
      for (int r = 0; r < 8; ++r)
        sc[wave * 16 + r + 8 * half][s0 + (lane & 15)] = acc[r];
    }
  }
  __syncthreads();

  if (tid < T_) {
    float mx = -3.4e38f;
    for (int s = 0; s < TENC_; ++s) mx = fmaxf(mx, sc[tid][s]);
    float sum = 0.f;
    for (int s = 0; s < TENC_; ++s) { float e = __expf(sc[tid][s] - mx); sc[tid][s] = e; sum += e; }
    float inv = 1.0f / sum;
    for (int s = 0; s < TENC_; ++s) sc[tid][s] *= inv;
  }
  __syncthreads();

  // ctx = attn @ enc  (M=T, N=U, K=T_ENC)
  {
    const float* arow = &sc[wave * 16 + (lane & 15)][0];
    v16bf a0 = ld_a_row(arow, half);
    v16bf a1 = ld_a_row(arow + 32, half);
    v16bf a2 = ld_a_row(arow + 64, half);
    v16bf a3 = ld_a_row(arow + 96, half);
    for (int n0 = 0; n0 < U_; n0 += 16) {
      v8f acc = {};
      acc = wmma_bf16(a0, ld_b_strided(encB + 0  * U_ + n0, U_, lane), acc);
      acc = wmma_bf16(a1, ld_b_strided(encB + 32 * U_ + n0, U_, lane), acc);
      acc = wmma_bf16(a2, ld_b_strided(encB + 64 * U_ + n0, U_, lane), acc);
      acc = wmma_bf16(a3, ld_b_strided(encB + 96 * U_ + n0, U_, lane), acc);
#pragma unroll
      for (int r = 0; r < 8; ++r) {
        int m = wave * 16 + r + 8 * half;
        ctx[((long)b * T_ + m) * U_ + n0 + (lane & 15)] = acc[r];
      }
    }
  }
}

// ---------------- Kernel 4: logits = [seq|ctx] @ Wd + bd  (M=4096,N=32000,K=128)
// B-fragments come pre-packed (bf16, fragment order): 2x b128 per fragment.
__global__ void __launch_bounds__(256) k_logits(
    const float* __restrict__ seq, const float* __restrict__ ctx,
    const __bf16* __restrict__ pw, const float* __restrict__ bd,
    float* __restrict__ out) {
  __shared__ float ft[16][2 * U_];         // 8 KB feat tile shared by 8 waves
  int tid  = threadIdx.x;
  int lane = tid & 31, wave = tid >> 5;
  int half = lane >> 4;
  int tileM = blockIdx.y * 16;

  for (int idx = tid; idx < 16 * 2 * U_; idx += 256) {
    int r = idx >> 7, k = idx & 127;
    int m = tileM + r;
    ft[r][k] = (k < U_) ? seq[(long)m * U_ + k] : ctx[(long)m * U_ + (k - U_)];
  }
  __syncthreads();

  const float* arow = &ft[lane & 15][0];
  v16bf a0 = ld_a_row(arow, half);
  v16bf a1 = ld_a_row(arow + 32, half);
  v16bf a2 = ld_a_row(arow + 64, half);
  v16bf a3 = ld_a_row(arow + 96, half);

  const v16bf* pwv = (const v16bf*)pw;

#pragma unroll
  for (int u = 0; u < 2; ++u) {
    int n0 = blockIdx.x * 256 + wave * 32 + u * 16;
    int nt = n0 >> 4;
    long fb = ((long)nt * 4) * 32 + lane;
    __builtin_prefetch(pwv + fb + 8 * 32, 0, 1);   // global_prefetch_b8: next tiles
    v16bf b0 = pwv[fb + 0 * 32];
    v16bf b1 = pwv[fb + 1 * 32];
    v16bf b2 = pwv[fb + 2 * 32];
    v16bf b3 = pwv[fb + 3 * 32];
    v8f acc = {};
    acc = wmma_bf16(a0, b0, acc);
    acc = wmma_bf16(a1, b1, acc);
    acc = wmma_bf16(a2, b2, acc);
    acc = wmma_bf16(a3, b3, acc);
    int n = n0 + (lane & 15);
    float bn = bd[n];
#pragma unroll
    for (int r = 0; r < 8; ++r) {
      int m = tileM + r + 8 * half;
      out[(long)m * V_ + n] = acc[r] + bn;
    }
  }
}

// ---------------- Kernel 5: in-place row softmax over V=32000 -------------------
__global__ void __launch_bounds__(256) k_softmax(float* __restrict__ out) {
  __shared__ float red[256];
  float* base = out + (long)blockIdx.x * V_;
  int tid = threadIdx.x;

  float mx = -3.4e38f;
  for (int i = tid; i < V_; i += 256) mx = fmaxf(mx, base[i]);
  red[tid] = mx; __syncthreads();
  for (int s = 128; s > 0; s >>= 1) {
    if (tid < s) red[tid] = fmaxf(red[tid], red[tid + s]);
    __syncthreads();
  }
  mx = red[0]; __syncthreads();

  float sum = 0.f;
  for (int i = tid; i < V_; i += 256) { float e = __expf(base[i] - mx); base[i] = e; sum += e; }
  red[tid] = sum; __syncthreads();
  for (int s = 128; s > 0; s >>= 1) {
    if (tid < s) red[tid] += red[tid + s];
    __syncthreads();
  }
  float inv = 1.0f / red[0];
  for (int i = tid; i < V_; i += 256) base[i] *= inv;
}

// ---------------- Host launch ----------------------------------------------------
extern "C" void kernel_launch(void* const* d_in, const int* in_sizes, int n_in,
                              void* d_out, int out_size, void* d_ws, size_t ws_size,
                              hipStream_t stream) {
  const float* enc  = (const float*)d_in[0];
  const int*   dec  = (const int*)  d_in[1];
  const float* h0   = (const float*)d_in[2];
  const float* c0   = (const float*)d_in[3];
  const float* emb  = (const float*)d_in[4];
  const float* Wx   = (const float*)d_in[5];
  const float* Wh   = (const float*)d_in[6];
  const float* bias = (const float*)d_in[7];
  const float* Wd   = (const float*)d_in[8];
  const float* bd   = (const float*)d_in[9];

  float* outF = (float*)d_out;
  float* xz   = (float*)d_ws;                      // 4096*256 f32 = 4 MB
  float* seq  = xz  + (size_t)M_ALL * G_;          // 1 MB
  float* ctx  = seq + (size_t)M_ALL * U_;          // 1 MB
  __bf16* pw  = (__bf16*)(ctx + (size_t)M_ALL * U_); // 2000*4*32*16 bf16 = 8.2 MB

  float* hT = outF + (size_t)M_ALL * V_;
  float* cT = hT + (size_t)B_ * U_;

  k_pack_wd   <<<(NT_ * 4 * 32) / 256, 256, 0, stream>>>(Wd, pw);
  k_embed_gemm<<<dim3(2, 256),   256,  0, stream>>>(dec, emb, Wx, bias, xz);
  k_lstm      <<<1,              1024, 0, stream>>>(xz, Wh, h0, c0, seq, hT, cT);
  k_attn      <<<B_,             256,  0, stream>>>(enc, h0, seq, ctx);
  k_logits    <<<dim3(125, 256), 256,  0, stream>>>(seq, ctx, pw, bd, outF);
  k_softmax   <<<M_ALL,          256,  0, stream>>>(outF);
}
